// FourierBlock_75557064671327
// MI455X (gfx1250) — matrix operand order repaired and checked by reading, not compiled, not run on hardware
//
#include <hip/hip_runtime.h>
#include <hip/hip_bf16.h>
#include <math.h>

typedef __attribute__((ext_vector_type(16))) _Float16 v16h;
typedef __attribute__((ext_vector_type(8)))  _Float16 v8h;
typedef __attribute__((ext_vector_type(8)))  float    v8f;
typedef __attribute__((ext_vector_type(4)))  int      v4i;

#define BB 32
#define LL 1024
#define EE 512
#define MM 64
#define W_SCALE 65536.0f   // lift W out of f16-subnormal range; undone in final f32 scale
#define PI_F 3.14159265358979323846f

// ---------------- CDNA5 async global->LDS copy (ASYNCcnt path) ----------------
#if defined(__has_builtin)
#if __has_builtin(__builtin_amdgcn_global_load_async_to_lds_b128)
#define FB_ASYNC 1
#endif
#endif

__device__ __forceinline__ void async_copy16(const _Float16* g, _Float16* l) {
#ifdef FB_ASYNC
  __builtin_amdgcn_global_load_async_to_lds_b128(
      (__attribute__((address_space(1))) v4i*)(uintptr_t)g,
      (__attribute__((address_space(3))) v4i*)(uintptr_t)l,
      0, 0);
#else
  *(v8h*)l = *(const v8h*)g;   // fallback: global_load + ds_store
#endif
}

template <int N>
__device__ __forceinline__ void async_wait_stage() {
#ifdef FB_ASYNC
#if __has_builtin(__builtin_amdgcn_s_wait_asynccnt)
  __builtin_amdgcn_s_wait_asynccnt(N);
#else
  asm volatile("s_wait_asynccnt %0" ::"i"(N) : "memory");
#endif
#endif
}

// ---------------- WMMA helpers (gfx1250 layouts per CDNA5 ISA 7.12.2) ----------------

__device__ __forceinline__ v8f wmma16(v16h a, v16h b, v8f c) {
  // v_wmma_f32_16x16x32_f16 : D = A(16x32 f16) * B(32x16 f16) + C(16x16 f32)
  return __builtin_amdgcn_wmma_f32_16x16x32_f16(false, a, false, b, (short)0, c, false, false);
}

// A-matrix (16-bit, 16x32): lane M = lane&15; elements 0..7 -> K = half*8 + e,
// elements 8..15 -> K = 16 + half*8 + (e-8), half = lane>>4.
__device__ __forceinline__ v16h load_A16(const _Float16* rowbase, int k0, int halfsel) {
  v8h lo = *(const v8h*)(rowbase + k0 + halfsel * 8);
  v8h hi = *(const v8h*)(rowbase + k0 + 16 + halfsel * 8);
  v16h r;
#pragma unroll
  for (int e = 0; e < 8; ++e) { r[e] = lo[e]; r[e + 8] = hi[e]; }
  return r;
}

// ---------------- Kernel 1: DFT coefficient tables (f16) ----------------
__global__ void fb_coef_kernel(_Float16* Dc, _Float16* Ds, _Float16* iC, _Float16* iS) {
  int idx = blockIdx.x * blockDim.x + threadIdx.x;  // 64*1024 threads
  int m = idx >> 10;
  int l = idx & (LL - 1);
  int ph = (m * l) & (LL - 1);                      // phase reduced mod L for accuracy
  float th = (2.0f * PI_F / (float)LL) * (float)ph;
  float c = cosf(th), s = sinf(th);
  Dc[m * LL + l] = (_Float16)c;                     // forward:  cos
  Ds[m * LL + l] = (_Float16)(-s);                  // forward: -sin (e^{-i theta})
  float w = (m == 0) ? 0.5f : 1.0f;
  iC[l * MM + m] = (_Float16)(w * c);               // inverse:  w*cos
  iS[l * MM + m] = (_Float16)(-w * s);              // inverse: -w*sin
}

// ---------------- Kernel 2: W transpose+convert: w[h][i][m] f32 -> Wt[m][h][i] f16*2^16 ----
__global__ void fb_wtrans_kernel(const float* __restrict__ wr, const float* __restrict__ wi,
                                 _Float16* __restrict__ WtR, _Float16* __restrict__ WtI) {
  __shared__ float tile[64][65];
  int h  = blockIdx.x >> 3;
  int i0 = (blockIdx.x & 7) * 64;
  int t  = threadIdx.x;
  for (int which = 0; which < 2; ++which) {
    const float* src = which ? wi : wr;
    _Float16*    dst = which ? WtI : WtR;
#pragma unroll
    for (int rep = 0; rep < 16; ++rep) {
      int idx = rep * 256 + t;
      int il = idx >> 6;            // 0..63
      int m  = idx & 63;            // consecutive t -> consecutive m: coalesced read
      tile[m][il] = src[((size_t)h * EE + (i0 + il)) * MM + m];
    }
    __syncthreads();
#pragma unroll
    for (int rep = 0; rep < 16; ++rep) {
      int idx = rep * 256 + t;
      int m  = idx >> 6;
      int il = idx & 63;            // consecutive t -> consecutive i: coalesced write
      dst[((size_t)m * EE + h) * EE + (i0 + il)] = (_Float16)(tile[m][il] * W_SCALE);
    }
    __syncthreads();
  }
}

// ---------------- Kernel 3: forward truncated DFT via WMMA ----------------
// Xg[m][b][h] = sum_l q[b][l][h] * (Dc + i*Ds)[m][l]
// A-panels (Dc/Ds k-chunks) are block-uniform -> async-staged to LDS, double-buffered.
__global__ void fb_fdft_kernel(const float* __restrict__ q,
                               const _Float16* __restrict__ Dc, const _Float16* __restrict__ Ds,
                               _Float16* __restrict__ XgR, _Float16* __restrict__ XgI) {
  __shared__ __align__(16) _Float16 sDc[2][64][40];  // 32 data + 8 pad halves per row
  __shared__ __align__(16) _Float16 sDs[2][64][40];

  int tid  = threadIdx.x;
  int wave = tid >> 5;
  int lane = tid & 31;
  int b  = blockIdx.x >> 2;                 // 0..31
  int hg = blockIdx.x & 3;                  // 0..3
  int h0 = (hg * 8 + wave) * 16;            // this wave's h-tile
  int halfsel = lane >> 4;
  int Mloc = lane & 15;

  // staging slot for this thread: one 16B chunk of Dc and one of Ds per k-step
  int sm   = tid >> 2;                      // row (mode) 0..63
  int sseg = (tid & 3) * 8;                 // col offset in halves (0,8,16,24)

  v8f accR[4] = {}, accI[4] = {};
  const float* qb = q + (size_t)b * LL * EE;

  // prologue: stage k0 = 0 into buffer 0
  async_copy16(Dc + (size_t)sm * LL + 0 + sseg, &sDc[0][sm][sseg]);
  async_copy16(Ds + (size_t)sm * LL + 0 + sseg, &sDs[0][sm][sseg]);

  for (int k0 = 0; k0 < LL; k0 += 32) {
    int buf = (k0 >> 5) & 1;
    if (k0 + 32 < LL) {                     // prefetch next panel into other buffer
      async_copy16(Dc + (size_t)sm * LL + (k0 + 32) + sseg, &sDc[buf ^ 1][sm][sseg]);
      async_copy16(Ds + (size_t)sm * LL + (k0 + 32) + sseg, &sDs[buf ^ 1][sm][sseg]);
    }
    async_wait_stage<2>();                  // current panel's 2 copies complete
    __syncthreads();                        // panel visible block-wide

    // B tile: lane -> N column (h = h0+Mloc), elements -> K in table order; f32->f16
    const float* brow = qb + (size_t)k0 * EE + h0 + Mloc;
    v16h Bq;
#pragma unroll
    for (int e = 0; e < 16; ++e) Bq[e] = (_Float16)brow[(size_t)(((e & 8) << 1) + halfsel * 8 + (e & 7)) * EE];

#pragma unroll
    for (int t = 0; t < 4; ++t) {           // 4 mode-tiles of 16 -> all 64 modes
      int m = t * 16 + Mloc;
      v16h Ac = load_A16(&sDc[buf][m][0], 0, halfsel);
      v16h As = load_A16(&sDs[buf][m][0], 0, halfsel);
      accR[t] = wmma16(Ac, Bq, accR[t]);
      accI[t] = wmma16(As, Bq, accI[t]);
    }
    __syncthreads();                        // done reading buf before it is re-staged
  }
#pragma unroll
  for (int t = 0; t < 4; ++t) {
#pragma unroll
    for (int r = 0; r < 8; ++r) {
      int m = t * 16 + r + 8 * halfsel;
      size_t o = ((size_t)m * BB + b) * EE + h0 + Mloc;
      XgR[o] = (_Float16)accR[t][r];
      XgI[o] = (_Float16)accI[t][r];
    }
  }
}

// ---------------- Kernel 4: per-mode complex mode mixing via WMMA ----------------
// Out[b][m][i] = sum_h Xg[m][b][h] * Wt[m][h][i]  (complex; stored *2^16)
// A-panels (Xg[m] k-chunks) are block-uniform -> async-staged to LDS, double-buffered.
__global__ void fb_mix_kernel(const _Float16* __restrict__ XgR, const _Float16* __restrict__ XgI,
                              const _Float16* __restrict__ WtR, const _Float16* __restrict__ WtI,
                              _Float16* __restrict__ OutR, _Float16* __restrict__ OutI) {
  __shared__ __align__(16) _Float16 sXr[2][32][40];
  __shared__ __align__(16) _Float16 sXi[2][32][40];

  int tid  = threadIdx.x;
  int wid  = blockIdx.x * 8 + (tid >> 5);          // 0..2047
  int lane = tid & 31;
  int m  = wid >> 5;                               // 0..63 (uniform per block)
  int i0 = (wid & 31) * 16;
  int halfsel = lane >> 4;
  int Mloc = lane & 15;

  // staging slot: 256 chunks (2 arrays * 32 rows * 4 segs) -> 1 per thread
  int sarr = tid >> 7;                             // 0: Xr, 1: Xi
  int srow = (tid >> 2) & 31;                      // b row
  int sseg = (tid & 3) * 8;

  v8f aRR[2] = {}, aII[2] = {}, aIm[2] = {};
  const _Float16* Xr = XgR + (size_t)m * BB * EE;
  const _Float16* Xi = XgI + (size_t)m * BB * EE;
  const _Float16* Wr = WtR + (size_t)m * EE * EE;
  const _Float16* Wi = WtI + (size_t)m * EE * EE;
  const _Float16* sg = (sarr ? Xi : Xr) + (size_t)srow * EE + sseg;

  async_copy16(sg + 0, (sarr ? &sXi[0][srow][sseg] : &sXr[0][srow][sseg]));

  for (int k0 = 0; k0 < EE; k0 += 32) {
    int buf = (k0 >> 5) & 1;
    if (k0 + 32 < EE)
      async_copy16(sg + (k0 + 32),
                   (sarr ? &sXi[buf ^ 1][srow][sseg] : &sXr[buf ^ 1][srow][sseg]));
    async_wait_stage<1>();
    __syncthreads();

    v16h Br, Bi;
#pragma unroll
    for (int e = 0; e < 16; ++e) {
      size_t o = (size_t)(k0 + ((e & 8) << 1) + halfsel * 8 + (e & 7)) * EE + i0 + Mloc;
      Br[e] = Wr[o]; Bi[e] = Wi[o];
    }
#pragma unroll
    for (int mt = 0; mt < 2; ++mt) {               // two b-tiles cover B=32
      int brow = mt * 16 + Mloc;
      v16h Ar = load_A16(&sXr[buf][brow][0], 0, halfsel);
      v16h Ai = load_A16(&sXi[buf][brow][0], 0, halfsel);
      aRR[mt] = wmma16(Ar, Br, aRR[mt]);           // Xr*Wr
      aII[mt] = wmma16(Ai, Bi, aII[mt]);           // Xi*Wi
      aIm[mt] = wmma16(Ar, Bi, aIm[mt]);           // Xr*Wi
      aIm[mt] = wmma16(Ai, Br, aIm[mt]);           // + Xi*Wr
    }
    __syncthreads();
  }
#pragma unroll
  for (int mt = 0; mt < 2; ++mt) {
#pragma unroll
    for (int r = 0; r < 8; ++r) {
      int b = mt * 16 + r + 8 * halfsel;
      size_t o = ((size_t)b * MM + m) * EE + i0 + Mloc;
      OutR[o] = (_Float16)(aRR[mt][r] - aII[mt][r]);  // real: XrWr - XiWi
      OutI[o] = (_Float16)(aIm[mt][r]);
    }
  }
}

// ---------------- Kernel 5: truncated inverse rDFT via WMMA, f32 output ----------------
// out[b][l][i] = (2/(L*W_SCALE)) * sum_m ( iC[l][m]*OutR[b][m][i] + iS[l][m]*OutI[b][m][i] )
// A-panel (iC/iS rows for this block's 64 l values, all 64 K) staged once to LDS.
__global__ void fb_idft_kernel(const _Float16* __restrict__ OutR, const _Float16* __restrict__ OutI,
                               const _Float16* __restrict__ iC, const _Float16* __restrict__ iS,
                               float* __restrict__ out) {
  __shared__ __align__(16) _Float16 sIC[64][72];   // 64 K + 8 pad halves per row
  __shared__ __align__(16) _Float16 sIS[64][72];

  int tid  = threadIdx.x;
  int wid  = blockIdx.x * 8 + (tid >> 5);          // 0..16383
  int lane = tid & 31;
  int i0     = (wid & 31) * 16;
  int lgroup = (wid >> 5) & 15;                    // uniform per block
  int b      = wid >> 9;                           // uniform per block
  int halfsel = lane >> 4;
  int Mloc = lane & 15;
  int lbase = lgroup * 64;

  // stage 1024 16B-chunks (2 arrays * 64 rows * 8 segs) with 256 threads: 4 each
#pragma unroll
  for (int c = 0; c < 4; ++c) {
    int idx  = c * 256 + tid;
    int arr  = idx >> 9;
    int row  = (idx >> 3) & 63;
    int seg  = (idx & 7) * 8;
    const _Float16* src = (arr ? iS : iC) + (size_t)(lbase + row) * MM + seg;
    async_copy16(src, (arr ? &sIS[row][seg] : &sIC[row][seg]));
  }
  async_wait_stage<0>();
  __syncthreads();

  v8f acc[4] = {};
  const _Float16* Rb = OutR + (size_t)b * MM * EE;
  const _Float16* Ib = OutI + (size_t)b * MM * EE;

#pragma unroll
  for (int k0 = 0; k0 < MM; k0 += 32) {            // K = 64 modes
    v16h BR, BI;
#pragma unroll
    for (int e = 0; e < 16; ++e) {
      size_t o = (size_t)(k0 + ((e & 8) << 1) + halfsel * 8 + (e & 7)) * EE + i0 + Mloc;
      BR[e] = Rb[o]; BI[e] = Ib[o];
    }
#pragma unroll
    for (int lt = 0; lt < 4; ++lt) {               // 4 l-tiles reuse the B load
      int lloc = lt * 16 + Mloc;
      v16h Ac = load_A16(&sIC[lloc][0], k0, halfsel);
      v16h As = load_A16(&sIS[lloc][0], k0, halfsel);
      acc[lt] = wmma16(Ac, BR, acc[lt]);
      acc[lt] = wmma16(As, BI, acc[lt]);
    }
  }
  const float FS = 2.0f / ((float)LL * W_SCALE);
#pragma unroll
  for (int lt = 0; lt < 4; ++lt) {
#pragma unroll
    for (int r = 0; r < 8; ++r) {
      int l = lbase + lt * 16 + r + 8 * halfsel;
      out[((size_t)b * LL + l) * EE + i0 + Mloc] = acc[lt][r] * FS;
    }
  }
}

// ---------------- launch ----------------
extern "C" void kernel_launch(void* const* d_in, const int* in_sizes, int n_in,
                              void* d_out, int out_size, void* d_ws, size_t ws_size,
                              hipStream_t stream) {
  const float* q  = (const float*)d_in[0];   // (B, L, E) f32
  const float* wr = (const float*)d_in[1];   // (E, E, M) f32
  const float* wi = (const float*)d_in[2];   // (E, E, M) f32
  float* out = (float*)d_out;                // (B, L, E) f32

  char* ws = (char*)d_ws;
  size_t off = 0;
  auto alloc = [&](size_t bytes) -> char* {
    char* p = ws + off;
    off = (off + bytes + 255) & ~(size_t)255;
    return p;
  };
  _Float16* Dc   = (_Float16*)alloc((size_t)MM * LL * 2);       // 128 KB
  _Float16* Ds   = (_Float16*)alloc((size_t)MM * LL * 2);       // 128 KB
  _Float16* iC   = (_Float16*)alloc((size_t)LL * MM * 2);       // 128 KB
  _Float16* iS   = (_Float16*)alloc((size_t)LL * MM * 2);       // 128 KB
  _Float16* XgR  = (_Float16*)alloc((size_t)MM * BB * EE * 2);  // 2 MB
  _Float16* XgI  = (_Float16*)alloc((size_t)MM * BB * EE * 2);  // 2 MB
  _Float16* OutR = (_Float16*)alloc((size_t)BB * MM * EE * 2);  // 2 MB
  _Float16* OutI = (_Float16*)alloc((size_t)BB * MM * EE * 2);  // 2 MB
  _Float16* WtR  = (_Float16*)alloc((size_t)MM * EE * EE * 2);  // 32 MB
  _Float16* WtI  = (_Float16*)alloc((size_t)MM * EE * EE * 2);  // 32 MB

  fb_coef_kernel  <<<(MM * LL) / 256, 256, 0, stream>>>(Dc, Ds, iC, iS);
  fb_wtrans_kernel<<<EE * 8,          256, 0, stream>>>(wr, wi, WtR, WtI);
  fb_fdft_kernel  <<<BB * 4,          256, 0, stream>>>(q, Dc, Ds, XgR, XgI);
  fb_mix_kernel   <<<(MM * 32) / 8,   256, 0, stream>>>(XgR, XgI, WtR, WtI, OutR, OutI);
  fb_idft_kernel  <<<(BB * 16 * 32) / 8, 256, 0, stream>>>(OutR, OutI, iC, iS, out);
}
